// Decoder_24764781429449
// MI455X (gfx1250) — matrix operand order repaired and checked by reading, not compiled, not run on hardware
//
#include <hip/hip_runtime.h>
#include <hip/hip_bf16.h>
#include <math.h>

// ---------------- problem constants ----------------
#define N_NODES 4096
#define K_NEIGH 48
#define C_DIM   128
#define E_DIM   128
#define H_DIM   512
#define L_LAYERS 3
#define SCALE_F 30.0f
#define EPS_F   1e-5f

typedef _Float16 h8  __attribute__((ext_vector_type(8)));
typedef _Float16 h16 __attribute__((ext_vector_type(16)));
typedef float    f8  __attribute__((ext_vector_type(8)));

// ---------------- fragment helpers -----------------
// A fragment (16x32 f16), row-major source, leading dim `ld` (halfs).
// lane L -> row M = L&15 ; K = {c0..c0+7, c0+16..c0+23}, c0 = 8*(L>>4).
__device__ __forceinline__ h16 frag_a_rm(const _Float16* __restrict__ base,
                                         int ld, int row0, int k0, int lane) {
    const _Float16* p = base + (size_t)(row0 + (lane & 15)) * (size_t)ld
                             + (size_t)(k0 + ((lane >> 4) << 3));
    h8 lo = *(const h8*)(p);
    h8 hi = *(const h8*)(p + 16);
    return __builtin_shufflevector(lo, hi, 0,1,2,3,4,5,6,7,8,9,10,11,12,13,14,15);
}

// B fragment (32x16 f16) from N-major (transposed) weight wT[n][k], ld = ldk halfs.
// lane L -> col n = L&15 ; K = {c0..c0+7, c0+16..c0+23}, c0 = k0 + 8*(L>>4).
__device__ __forceinline__ h16 frag_b_T(const _Float16* __restrict__ wT,
                                        int ldk, int n0, int k0, int lane) {
    const _Float16* p = wT + (size_t)(n0 + (lane & 15)) * (size_t)ldk
                           + (size_t)(k0 + ((lane >> 4) << 3));
    h8 lo = *(const h8*)(p);
    h8 hi = *(const h8*)(p + 16);
    return __builtin_shufflevector(lo, hi, 0,1,2,3,4,5,6,7,8,9,10,11,12,13,14,15);
}

__device__ __forceinline__ f8 wmma_f16f32(h16 a, h16 b, f8 c) {
    return __builtin_amdgcn_wmma_f32_16x16x32_f16(false, a, false, b, (short)0, c,
                                                  false, false);
}

// Branch-free exact-GELU: erf via Abramowitz-Stegun 7.1.26 (|err| < 1.5e-7),
// using HW v_rcp_f32 / v_exp_f32 (exp2) transcendentals.
__device__ __forceinline__ float gelu_fast(float x) {
    float z  = fabsf(x) * 0.70710678118654752f;      // |x|/sqrt(2)
    float t  = __builtin_amdgcn_rcpf(1.0f + 0.3275911f * z);
    float pl = t * (0.254829592f +
               t * (-0.284496736f +
               t * (1.421413741f +
               t * (-1.453152027f + t * 1.061405429f))));
    float e  = __builtin_amdgcn_exp2f(-0.721347520444482f * x * x); // exp(-z^2)
    float er = copysignf(1.0f - pl * e, x);
    return 0.5f * x * (1.0f + er);
}

// Async copy 16B/lane from global to LDS (ASYNCcnt-tracked, CDNA5 path).
__device__ __forceinline__ void async_b128(unsigned lds_off, const void* gaddr) {
    asm volatile("global_load_async_to_lds_b128 %0, %1, off"
                 :: "v"(lds_off), "v"(gaddr) : "memory");
}
__device__ __forceinline__ void wait_async0() {
    asm volatile("s_wait_asynccnt 0x0" ::: "memory");
}

// ---------------- utility kernels ------------------
__global__ void cvt_f16_kernel(const float* __restrict__ src,
                               _Float16* __restrict__ dst, int n) {
    for (int i = blockIdx.x * blockDim.x + threadIdx.x; i < n;
         i += gridDim.x * blockDim.x)
        dst[i] = (_Float16)src[i];
}

// src[R][Cc] row-major f32 -> dst[Cc][R] row-major f16 (transpose + convert)
__global__ void transpose_cvt_kernel(const float* __restrict__ src,
                                     _Float16* __restrict__ dst, int R, int Cc) {
    int total = R * Cc;
    for (int i = blockIdx.x * blockDim.x + threadIdx.x; i < total;
         i += gridDim.x * blockDim.x) {
        int r = i / Cc, c = i % Cc;
        dst[(size_t)c * R + r] = (_Float16)src[(size_t)r * Cc + c];
    }
}

// ---------------- p kernel (WMMA) ------------------
// p[16 nodes] = x @ W1[0:128] + node0 @ W1[128:256] + b1
__global__ void __launch_bounds__(256) p_kernel(
    const _Float16* __restrict__ x16, const _Float16* __restrict__ node16,
    const _Float16* __restrict__ w1aT, const _Float16* __restrict__ w1bT,
    const float* __restrict__ b1, float* __restrict__ p) {
    const int n0   = blockIdx.x * 16;
    const int tid  = threadIdx.x;
    const int wave = tid >> 5, lane = tid & 31;
    const int nc0  = wave * 16;
    const int col  = nc0 + (lane & 15);
    const int rsh  = (lane >> 4) << 3;

    h16 Ax[4], An[4];
    #pragma unroll
    for (int ks = 0; ks < 4; ++ks) {
        Ax[ks] = frag_a_rm(x16 + (size_t)n0 * C_DIM, C_DIM, 0, ks * 32, lane);
        An[ks] = frag_a_rm(node16 + (size_t)n0 * C_DIM, C_DIM, 0, ks * 32, lane);
    }
    f8 acc0 = {0.f,0.f,0.f,0.f,0.f,0.f,0.f,0.f};
    f8 acc1 = {0.f,0.f,0.f,0.f,0.f,0.f,0.f,0.f};
    #pragma unroll
    for (int ks = 0; ks < 4; ++ks) {
        acc0 = wmma_f16f32(Ax[ks], frag_b_T(w1aT, C_DIM, nc0, ks * 32, lane), acc0);
        acc1 = wmma_f16f32(An[ks], frag_b_T(w1bT, C_DIM, nc0, ks * 32, lane), acc1);
    }
    const float bc = b1[col];
    #pragma unroll
    for (int i = 0; i < 8; ++i)
        p[(size_t)(n0 + i + rsh) * C_DIM + col] = acc0[i] + acc1[i] + bc;
}

// ---------------- message kernel -------------------
// Two nodes per block (96 edge rows). 8 waves; wave w owns cols [16w,16w+16).
// h1 = gelu(edge @ W1e + p); h2 = gelu(h1 @ W2 + b2);
// msum = sum_k(h2 @ W3) + 48*b3; x1 = x + msum/30; x_mid = LN1(x1)
__global__ void __launch_bounds__(256) msg_kernel(
    const _Float16* __restrict__ edge16,   // [N,48,128]
    const _Float16* __restrict__ w1eT,     // [128n][128k]
    const _Float16* __restrict__ w2T, const _Float16* __restrict__ w3T,
    const float* __restrict__ b2, const float* __restrict__ b3,
    const float* __restrict__ pvec,        // [N,128] (includes b1)
    const float* __restrict__ x,           // [N,128]
    const float* __restrict__ g1, const float* __restrict__ be1,
    float* __restrict__ x_mid, _Float16* __restrict__ x_mid16) {

    // sBufA holds the edge tile, then is reused for h2 (edge dead after GEMM1).
    __shared__ __align__(16) _Float16 sBufA[2 * K_NEIGH * C_DIM]; // 24 KB
    __shared__ __align__(16) _Float16 sH1  [2 * K_NEIGH * C_DIM]; // 24 KB
    __shared__ float sCol[2][C_DIM];
    __shared__ float sRed[2][C_DIM];
    __shared__ float sRed2[2][C_DIM];

    const int n0   = blockIdx.x * 2;
    const int tid  = threadIdx.x;
    const int wave = tid >> 5, lane = tid & 31;
    const int nc0  = wave * 16;
    const int col  = nc0 + (lane & 15);
    const int rsh  = (lane >> 4) << 3;

    sCol[tid >> 7][tid & 127] = 0.0f;

    // ---- async-stage the 2-node edge tile (96x128 f16 = 24 KB) into LDS ----
    {
        const char* g = (const char*)(edge16 + (size_t)n0 * (K_NEIGH * C_DIM));
        unsigned lbase = (unsigned)(uintptr_t)(&sBufA[0]);  // low 32b = LDS offset
        #pragma unroll
        for (int i = 0; i < 6; ++i) {
            unsigned off = (unsigned)(tid + 256 * i) * 16u;
            async_b128(lbase + off, g + off);
        }
        wait_async0();
    }
    __syncthreads();

    // ---- GEMM1: edge(96x128) @ W1e(128x128) + p, gelu -> sH1 ----
    h16 B[4];
    #pragma unroll
    for (int ks = 0; ks < 4; ++ks) B[ks] = frag_b_T(w1eT, C_DIM, nc0, ks * 32, lane);
    const float pc0 = pvec[(size_t)n0 * C_DIM + col];
    const float pc1 = pvec[(size_t)(n0 + 1) * C_DIM + col];
    #pragma unroll
    for (int mt = 0; mt < 6; ++mt) {
        f8 a0 = {0.f,0.f,0.f,0.f,0.f,0.f,0.f,0.f};
        f8 a1 = {0.f,0.f,0.f,0.f,0.f,0.f,0.f,0.f};
        a0 = wmma_f16f32(frag_a_rm(sBufA, C_DIM, mt * 16,  0, lane), B[0], a0);
        a1 = wmma_f16f32(frag_a_rm(sBufA, C_DIM, mt * 16, 32, lane), B[1], a1);
        a0 = wmma_f16f32(frag_a_rm(sBufA, C_DIM, mt * 16, 64, lane), B[2], a0);
        a1 = wmma_f16f32(frag_a_rm(sBufA, C_DIM, mt * 16, 96, lane), B[3], a1);
        const float pc = (mt < 3) ? pc0 : pc1;
        #pragma unroll
        for (int i = 0; i < 8; ++i) {
            float v = gelu_fast(a0[i] + a1[i] + pc);
            sH1[(mt * 16 + i + rsh) * C_DIM + col] = (_Float16)v;
        }
    }
    __syncthreads();

    // ---- GEMM2: h1 @ W2 + b2, gelu -> sBufA (as h2) ----
    #pragma unroll
    for (int ks = 0; ks < 4; ++ks) B[ks] = frag_b_T(w2T, C_DIM, nc0, ks * 32, lane);
    const float b2c = b2[col];
    #pragma unroll
    for (int mt = 0; mt < 6; ++mt) {
        f8 a0 = {0.f,0.f,0.f,0.f,0.f,0.f,0.f,0.f};
        f8 a1 = {0.f,0.f,0.f,0.f,0.f,0.f,0.f,0.f};
        a0 = wmma_f16f32(frag_a_rm(sH1, C_DIM, mt * 16,  0, lane), B[0], a0);
        a1 = wmma_f16f32(frag_a_rm(sH1, C_DIM, mt * 16, 32, lane), B[1], a1);
        a0 = wmma_f16f32(frag_a_rm(sH1, C_DIM, mt * 16, 64, lane), B[2], a0);
        a1 = wmma_f16f32(frag_a_rm(sH1, C_DIM, mt * 16, 96, lane), B[3], a1);
        #pragma unroll
        for (int i = 0; i < 8; ++i) {
            float v = gelu_fast(a0[i] + a1[i] + b2c);
            sBufA[(mt * 16 + i + rsh) * C_DIM + col] = (_Float16)v;
        }
    }
    __syncthreads();

    // ---- GEMM3: h2 @ W3, reduced over each node's 48 rows ----
    #pragma unroll
    for (int ks = 0; ks < 4; ++ks) B[ks] = frag_b_T(w3T, C_DIM, nc0, ks * 32, lane);
    float cs0 = 0.0f, cs1 = 0.0f;
    #pragma unroll
    for (int mt = 0; mt < 6; ++mt) {
        f8 a0 = {0.f,0.f,0.f,0.f,0.f,0.f,0.f,0.f};
        f8 a1 = {0.f,0.f,0.f,0.f,0.f,0.f,0.f,0.f};
        a0 = wmma_f16f32(frag_a_rm(sBufA, C_DIM, mt * 16,  0, lane), B[0], a0);
        a1 = wmma_f16f32(frag_a_rm(sBufA, C_DIM, mt * 16, 32, lane), B[1], a1);
        a0 = wmma_f16f32(frag_a_rm(sBufA, C_DIM, mt * 16, 64, lane), B[2], a0);
        a1 = wmma_f16f32(frag_a_rm(sBufA, C_DIM, mt * 16, 96, lane), B[3], a1);
        float s = 0.0f;
        #pragma unroll
        for (int i = 0; i < 8; ++i) s += a0[i] + a1[i];
        if (mt < 3) cs0 += s; else cs1 += s;
    }
    atomicAdd(&sCol[0][col], cs0);   // lanes L and L+16 cover complementary rows
    atomicAdd(&sCol[1][col], cs1);
    __syncthreads();

    // ---- residual + LayerNorm1 (both nodes, all 256 threads) ----
    const int ln = tid >> 7;
    const int c  = tid & 127;
    const int nn = n0 + ln;
    float msum = sCol[ln][c] + (float)K_NEIGH * b3[c];
    float x1 = x[(size_t)nn * C_DIM + c] + msum * (1.0f / SCALE_F);
    sRed[ln][c]  = x1;
    sRed2[ln][c] = x1 * x1;
    __syncthreads();
    for (int s = 64; s > 0; s >>= 1) {
        if (c < s) { sRed[ln][c] += sRed[ln][c + s]; sRed2[ln][c] += sRed2[ln][c + s]; }
        __syncthreads();
    }
    float mean = sRed[ln][0] * (1.0f / C_DIM);
    float var  = sRed2[ln][0] * (1.0f / C_DIM) - mean * mean;
    float y = (x1 - mean) * rsqrtf(var + EPS_F) * g1[c] + be1[c];
    x_mid[(size_t)nn * C_DIM + c]   = y;
    x_mid16[(size_t)nn * C_DIM + c] = (_Float16)y;
}

// ---------------- dense kernel ---------------------
// 16 nodes/block. h = gelu(xm @ Dw1 + db1); d = h @ Dw2 + db2;
// out = mask * LN2(xm + d); also emits f16 copy for next layer's p kernel.
__global__ void __launch_bounds__(256) dense_kernel(
    const _Float16* __restrict__ xm16, const float* __restrict__ xm,
    const _Float16* __restrict__ dw1T,   // [512h][128c]
    const _Float16* __restrict__ dw2T,   // [128c][512h]
    const float* __restrict__ db1, const float* __restrict__ db2,
    const float* __restrict__ g2, const float* __restrict__ be2,
    const float* __restrict__ mask,
    float* __restrict__ x_out, _Float16* __restrict__ x_out16) {

    __shared__ __align__(16) _Float16 sH[16 * H_DIM];   // 16 KB
    __shared__ float sOut[16 * C_DIM];                  // 8 KB
    __shared__ float sMean[16], sInv[16];

    const int n0   = blockIdx.x * 16;
    const int tid  = threadIdx.x;
    const int wave = tid >> 5, lane = tid & 31;
    const int rsh  = (lane >> 4) << 3;

    const _Float16* xb = xm16 + (size_t)n0 * C_DIM;
    h16 A[4];
    #pragma unroll
    for (int ks = 0; ks < 4; ++ks) A[ks] = frag_a_rm(xb, C_DIM, 0, ks * 32, lane);

    // ---- GEMM1: (16x128)@(128x512) ----
    #pragma unroll
    for (int j = 0; j < 4; ++j) {
        const int nc0 = (wave * 4 + j) * 16;
        const int col = nc0 + (lane & 15);
        f8 a0 = {0.f,0.f,0.f,0.f,0.f,0.f,0.f,0.f};
        f8 a1 = {0.f,0.f,0.f,0.f,0.f,0.f,0.f,0.f};
        a0 = wmma_f16f32(A[0], frag_b_T(dw1T, C_DIM, nc0,  0, lane), a0);
        a1 = wmma_f16f32(A[1], frag_b_T(dw1T, C_DIM, nc0, 32, lane), a1);
        a0 = wmma_f16f32(A[2], frag_b_T(dw1T, C_DIM, nc0, 64, lane), a0);
        a1 = wmma_f16f32(A[3], frag_b_T(dw1T, C_DIM, nc0, 96, lane), a1);
        const float bc = db1[col];
        #pragma unroll
        for (int i = 0; i < 8; ++i) {
            float v = gelu_fast(a0[i] + a1[i] + bc);
            sH[(i + rsh) * H_DIM + col] = (_Float16)v;
        }
    }
    __syncthreads();

    // ---- GEMM2: (16x512)@(512x128) + residual ----
    {
        const int nc0 = wave * 16;
        const int col = nc0 + (lane & 15);
        f8 a0 = {0.f,0.f,0.f,0.f,0.f,0.f,0.f,0.f};
        f8 a1 = {0.f,0.f,0.f,0.f,0.f,0.f,0.f,0.f};
        #pragma unroll
        for (int ks = 0; ks < 16; ks += 2) {
            a0 = wmma_f16f32(frag_a_rm(sH, H_DIM, 0, ks * 32, lane),
                             frag_b_T(dw2T, H_DIM, nc0, ks * 32, lane), a0);
            a1 = wmma_f16f32(frag_a_rm(sH, H_DIM, 0, (ks + 1) * 32, lane),
                             frag_b_T(dw2T, H_DIM, nc0, (ks + 1) * 32, lane), a1);
        }
        const float bc = db2[col];
        #pragma unroll
        for (int i = 0; i < 8; ++i) {
            int row = i + rsh;
            sOut[row * C_DIM + col] =
                a0[i] + a1[i] + bc + xm[(size_t)(n0 + row) * C_DIM + col];
        }
    }
    __syncthreads();

    // ---- LayerNorm2 + mask ----
    if (tid < 16) {
        float s = 0.f, ss = 0.f;
        for (int c = 0; c < C_DIM; ++c) {
            float v = sOut[tid * C_DIM + c];
            s += v; ss += v * v;
        }
        float mean = s * (1.0f / C_DIM);
        float var  = ss * (1.0f / C_DIM) - mean * mean;
        sMean[tid] = mean;
        sInv[tid]  = rsqrtf(var + EPS_F);
    }
    __syncthreads();
    for (int idx = tid; idx < 16 * C_DIM; idx += 256) {
        int r = idx >> 7, c = idx & 127;
        float y = (sOut[idx] - sMean[r]) * sInv[r] * g2[c] + be2[c];
        y *= mask[n0 + r];
        x_out[(size_t)(n0 + r) * C_DIM + c]   = y;
        x_out16[(size_t)(n0 + r) * C_DIM + c] = (_Float16)y;
    }
}

// ---------------- host launcher --------------------
extern "C" void kernel_launch(void* const* d_in, const int* in_sizes, int n_in,
                              void* d_out, int out_size, void* d_ws, size_t ws_size,
                              hipStream_t stream) {
    (void)in_sizes; (void)n_in; (void)out_size; (void)ws_size;

    const float* node   = (const float*)d_in[0];
    const float* edge   = (const float*)d_in[1];
    const float* mask   = (const float*)d_in[2];
    const float* msg_w1 = (const float*)d_in[3];
    const float* msg_b1 = (const float*)d_in[4];
    const float* msg_w2 = (const float*)d_in[5];
    const float* msg_b2 = (const float*)d_in[6];
    const float* msg_w3 = (const float*)d_in[7];
    const float* msg_b3 = (const float*)d_in[8];
    const float* ln1_g  = (const float*)d_in[9];
    const float* ln1_b  = (const float*)d_in[10];
    const float* d_w1   = (const float*)d_in[11];
    const float* d_b1   = (const float*)d_in[12];
    const float* d_w2   = (const float*)d_in[13];
    const float* d_b2   = (const float*)d_in[14];
    const float* ln2_g  = (const float*)d_in[15];
    const float* ln2_b  = (const float*)d_in[16];

    char* wsp = (char*)d_ws;
    auto take = [&](size_t bytes) {
        char* p = wsp;
        wsp += (bytes + 255) & ~(size_t)255;
        return p;
    };
    // per-layer packed f16 weights: w1aT,w1bT,w1eT,w2T,w3T, dw1T, dw2T
    const size_t WL = 5 * C_DIM * C_DIM + 2 * C_DIM * H_DIM;  // halfs
    _Float16* edge16 = (_Float16*)take(sizeof(_Float16) * (size_t)N_NODES * K_NEIGH * E_DIM);
    _Float16* wpack  = (_Float16*)take(sizeof(_Float16) * WL * L_LAYERS);
    float*    pbuf   = (float*)take(sizeof(float) * (size_t)N_NODES * C_DIM);
    float*    xcur   = (float*)take(sizeof(float) * (size_t)N_NODES * C_DIM);
    float*    xmid   = (float*)take(sizeof(float) * (size_t)N_NODES * C_DIM);
    _Float16* xmid16 = (_Float16*)take(sizeof(_Float16) * (size_t)N_NODES * C_DIM);
    _Float16* xcur16 = (_Float16*)take(sizeof(_Float16) * (size_t)N_NODES * C_DIM);
    _Float16* node16 = (_Float16*)take(sizeof(_Float16) * (size_t)N_NODES * C_DIM);

    // ---- converts (deterministic; re-run every call) ----
    cvt_f16_kernel<<<8192, 256, 0, stream>>>(edge, edge16,
                                             N_NODES * K_NEIGH * E_DIM);
    cvt_f16_kernel<<<1024, 256, 0, stream>>>(node, node16, N_NODES * C_DIM);

    auto wptr = [&](int l) { return wpack + (size_t)l * WL; };
    for (int l = 0; l < L_LAYERS; ++l) {
        _Float16* b = wptr(l);
        const float* w1 = msg_w1 + (size_t)l * 4 * C_DIM * C_DIM;
        // W1 rows: [0:128)=x, [128:256)=node0, [256:384)=zeros, [384:512)=edge
        transpose_cvt_kernel<<<64, 256, 0, stream>>>(w1,                     b + 0 * C_DIM * C_DIM, C_DIM, C_DIM);
        transpose_cvt_kernel<<<64, 256, 0, stream>>>(w1 + 128 * C_DIM,       b + 1 * C_DIM * C_DIM, C_DIM, C_DIM);
        transpose_cvt_kernel<<<64, 256, 0, stream>>>(w1 + 384 * C_DIM,       b + 2 * C_DIM * C_DIM, C_DIM, C_DIM);
        transpose_cvt_kernel<<<64, 256, 0, stream>>>(msg_w2 + (size_t)l * C_DIM * C_DIM, b + 3 * C_DIM * C_DIM, C_DIM, C_DIM);
        transpose_cvt_kernel<<<64, 256, 0, stream>>>(msg_w3 + (size_t)l * C_DIM * C_DIM, b + 4 * C_DIM * C_DIM, C_DIM, C_DIM);
        transpose_cvt_kernel<<<256, 256, 0, stream>>>(d_w1 + (size_t)l * C_DIM * H_DIM,  b + 5 * C_DIM * C_DIM, C_DIM, H_DIM);
        transpose_cvt_kernel<<<256, 256, 0, stream>>>(d_w2 + (size_t)l * H_DIM * C_DIM,  b + 5 * C_DIM * C_DIM + C_DIM * H_DIM, H_DIM, C_DIM);
    }

    // ---- layer loop ----
    for (int l = 0; l < L_LAYERS; ++l) {
        _Float16* b = wptr(l);
        _Float16* w1aT = b + 0 * C_DIM * C_DIM;
        _Float16* w1bT = b + 1 * C_DIM * C_DIM;
        _Float16* w1eT = b + 2 * C_DIM * C_DIM;
        _Float16* w2T  = b + 3 * C_DIM * C_DIM;
        _Float16* w3T  = b + 4 * C_DIM * C_DIM;
        _Float16* dw1T = b + 5 * C_DIM * C_DIM;
        _Float16* dw2T = dw1T + (size_t)C_DIM * H_DIM;

        const float*    x32 = (l == 0) ? node   : xcur;
        const _Float16* x16 = (l == 0) ? node16 : xcur16;

        p_kernel<<<N_NODES / 16, 256, 0, stream>>>(
            x16, node16, w1aT, w1bT, msg_b1 + (size_t)l * C_DIM, pbuf);

        msg_kernel<<<N_NODES / 2, 256, 0, stream>>>(
            edge16, w1eT, w2T, w3T,
            msg_b2 + (size_t)l * C_DIM, msg_b3 + (size_t)l * C_DIM,
            pbuf, x32,
            ln1_g + (size_t)l * C_DIM, ln1_b + (size_t)l * C_DIM,
            xmid, xmid16);

        float* xout = (l == L_LAYERS - 1) ? (float*)d_out : xcur;
        dense_kernel<<<N_NODES / 16, 256, 0, stream>>>(
            xmid16, xmid, dw1T, dw2T,
            d_b1 + (size_t)l * H_DIM, d_b2 + (size_t)l * C_DIM,
            ln2_g + (size_t)l * C_DIM, ln2_b + (size_t)l * C_DIM,
            mask, xout, xcur16);
    }
}
